// PNAnet4L_21251498181120
// MI455X (gfx1250) — compile-verified, matrix operand order checked by reading, not run on hardware
//
#include <hip/hip_runtime.h>
#include <hip/hip_bf16.h>
#include <math.h>

#define NNODES 50000
#define NEDGES 800000
#define NGRAPH 128
#define FEAT   128
#define EPSV   1e-5f

typedef __attribute__((ext_vector_type(16))) __bf16 bf16x16;
typedef __attribute__((ext_vector_type(8)))  float  f32x8;

// ---------- monotonic float<->uint encoding for atomic min/max ----------
__device__ __forceinline__ unsigned encf(float f) {
    unsigned u = __float_as_uint(f);
    return (u & 0x80000000u) ? ~u : (u | 0x80000000u);
}
__device__ __forceinline__ float decf(unsigned u) {
    return __uint_as_float((u & 0x80000000u) ? (u & 0x7fffffffu) : ~u);
}

// ---------- init kernels ----------
__global__ void initf_kernel(float* p, float v, long long n) {
    size_t i = (size_t)blockIdx.x * blockDim.x + threadIdx.x;
    if (i < (size_t)n) p[i] = v;
}
__global__ void initu_kernel(unsigned* p, unsigned v, long long n) {
    size_t i = (size_t)blockIdx.x * blockDim.x + threadIdx.x;
    if (i < (size_t)n) p[i] = v;
}

// ---------- fold edge_encoder into pre_nn:  Wc = We @ We2 , bc = be@We2 + bpre ----------
__global__ void wc_kernel(const float* __restrict__ We, const float* __restrict__ be,
                          const float* __restrict__ Wpre, const float* __restrict__ bpre,
                          float* __restrict__ Wc, float* __restrict__ bcv, int f) {
    int j = threadIdx.x;
    if (j >= f) return;
    float wc0 = 0.f, wc1 = 0.f, wc2 = 0.f, wc3 = 0.f;
    float b = bpre[j];
    for (int k = 0; k < f; ++k) {
        float w2 = Wpre[(size_t)(2 * f + k) * f + j];   // We2[k][j]
        wc0 += We[0 * f + k] * w2;
        wc1 += We[1 * f + k] * w2;
        wc2 += We[2 * f + k] * w2;
        wc3 += We[3 * f + k] * w2;
        b   += be[k] * w2;
    }
    Wc[0 * f + j] = wc0; Wc[1 * f + j] = wc1; Wc[2 * f + j] = wc2; Wc[3 * f + j] = wc3;
    bcv[j] = b;
}

// ---------- pre-swizzle weight matrix into per-lane WMMA B-fragment layout (bf16) ----------
// Layout: [(kt*NT + nt)][lane 0..31][elem 0..15] contiguous; per the ISA 16-bit B layout,
// lane -> column n = nt*16 + lane%16, elem j -> k = kt*32 + (j<8 ? off+j : 16+off+j-8),
// off = (lane>=16)?8:0.  GEMM loads B fragments straight from global (WGP$/L2 broadcast).
__global__ void swizzlew_kernel(const float* __restrict__ W, __bf16* __restrict__ Wsw,
                                int K, int NT) {
    int i = blockIdx.x * blockDim.x + threadIdx.x;
    int total = K * NT * 16;            // == K * NCOLS
    if (i >= total) return;
    int j    = i & 15;
    int lane = (i >> 4) & 31;
    int blk  = i >> 9;                  // (kt*NT + nt)
    int nt   = blk % NT;
    int kt   = blk / NT;
    int off  = (lane >= 16) ? 8 : 0;
    int k    = kt * 32 + ((j < 8) ? (off + j) : (16 + off + (j - 8)));
    int n    = nt * 16 + (lane & 15);
    Wsw[i] = (__bf16)W[(size_t)k * (NT * 16) + n];
}

// ---------- WMMA bf16 GEMM:  C[M x NCOLS] (+)= act( rowscale.A @ W + bias ) ----------
// A: M x K (float or bf16, row-major, K % 32 == 0), Wsw: pre-swizzled bf16 B fragments.
// flags bit0: accumulate into C, bit1: relu epilogue.
template <typename AT, int NT>
__global__ __launch_bounds__(256) void gemm_wmma(const AT* __restrict__ A,
                                                 const __bf16* __restrict__ Wsw,
                                                 const float* __restrict__ bias,
                                                 const float* __restrict__ rowscale,
                                                 float* __restrict__ C,
                                                 int M, int K, int flags) {
    constexpr int NCOLS = NT * 16;
    __shared__ alignas(16) __bf16 ldsA[128][40];     // 128 rows x 32 K (bf16), 80B row pitch
    const int t     = threadIdx.x;
    const int m0    = blockIdx.x * 128;
    const int w     = t >> 5;           // wave id (8 waves of 32)
    const int lane  = t & 31;
    const int lmod  = lane & 15;
    const int lhalf = (lane >= 16) ? 1 : 0;
    const int koff  = lhalf ? 8 : 0;    // K offset per ISA 16-bit A layout

    f32x8 acc[NT];
#pragma unroll
    for (int i = 0; i < NT; ++i)
#pragma unroll
        for (int r = 0; r < 8; ++r) acc[i][r] = 0.f;

    for (int k0 = 0; k0 < K; k0 += 32) {
        __syncthreads();
        // A tile: 512 chunks of 8 contiguous k-elements; 2 chunks per thread, vectorized.
#pragma unroll
        for (int i = 0; i < 2; ++i) {
            int ch = t + i * 256;
            int r  = ch >> 2;           // 4 chunks per row
            int kc = (ch & 3) * 8;
            int m  = m0 + r;
            union { uint4 q; __bf16 h[8]; } ou;
            if (m < M) {
                const AT* ap = A + (size_t)m * K + k0 + kc;
                float rs = rowscale ? rowscale[m] : 1.f;
                if constexpr (sizeof(AT) == 4) {
                    const float4* fp = (const float4*)ap;
                    float4 a0 = fp[0], a1 = fp[1];
                    ou.h[0] = (__bf16)(a0.x * rs); ou.h[1] = (__bf16)(a0.y * rs);
                    ou.h[2] = (__bf16)(a0.z * rs); ou.h[3] = (__bf16)(a0.w * rs);
                    ou.h[4] = (__bf16)(a1.x * rs); ou.h[5] = (__bf16)(a1.y * rs);
                    ou.h[6] = (__bf16)(a1.z * rs); ou.h[7] = (__bf16)(a1.w * rs);
                } else {
                    union { uint4 q; __bf16 h[8]; } in;
                    in.q = *(const uint4*)ap;
                    if (rowscale) {
#pragma unroll
                        for (int jj = 0; jj < 8; ++jj) ou.h[jj] = (__bf16)((float)in.h[jj] * rs);
                    } else {
                        ou.q = in.q;
                    }
                }
                if (k0 + 32 < K)                     // prefetch next K tile of A
                    __builtin_prefetch(ap + 32, 0, 3);
            } else {
                ou.q = make_uint4(0u, 0u, 0u, 0u);
            }
            *(uint4*)&ldsA[r][kc] = ou.q;            // single ds_store_b128
        }
        __syncthreads();

        union { uint4 q[2]; bf16x16 v; } ua, ub;
        int mr = w * 16 + lmod;
        ua.q[0] = *(const uint4*)&ldsA[mr][koff];
        ua.q[1] = *(const uint4*)&ldsA[mr][16 + koff];
#pragma unroll
        for (int nt = 0; nt < NT; ++nt) {
            const uint4* bp =
                (const uint4*)(Wsw + (((size_t)(k0 >> 5) * NT + nt) * 32 + lane) * 16);
            ub.q[0] = bp[0];
            ub.q[1] = bp[1];
            acc[nt] = __builtin_amdgcn_wmma_f32_16x16x32_bf16(
                false, ua.v, false, ub.v, (short)0, acc[nt], false, false);
        }
    }
    // epilogue (C/D layout: vgpr r -> row r (+8 for upper half), lane%16 -> col)
#pragma unroll
    for (int nt = 0; nt < NT; ++nt) {
        int col = nt * 16 + lmod;
#pragma unroll
        for (int r = 0; r < 8; ++r) {
            int row = m0 + w * 16 + r + lhalf * 8;
            if (row < M) {
                float v = acc[nt][r];
                if (bias) v += bias[col];
                size_t idx = (size_t)row * NCOLS + col;
                if (flags & 1) v += C[idx];
                if (flags & 2) v = fmaxf(v, 0.f);
                C[idx] = v;
            }
        }
    }
}

// ---------- per-edge message tail t = Q[src] + edge_attr @ Wc, atomically aggregated at dst ----------
__global__ void edge_kernel(const int* __restrict__ src, const int* __restrict__ dst,
                            const float* __restrict__ EA, const float* __restrict__ Q,
                            const float* __restrict__ Wc,
                            float* __restrict__ S, float* __restrict__ SS,
                            unsigned* __restrict__ MN, unsigned* __restrict__ MX,
                            unsigned* __restrict__ DEG, int E, int f) {
    size_t i = (size_t)blockIdx.x * blockDim.x + threadIdx.x;
    if (i >= (size_t)E * f) return;
    int e = (int)(i / f);
    int j = (int)(i - (size_t)e * f);
    int s = src[e], d = dst[e];
    float t = Q[(size_t)s * f + j];
    float a0 = EA[(size_t)e * 4 + 0], a1 = EA[(size_t)e * 4 + 1];
    float a2 = EA[(size_t)e * 4 + 2], a3 = EA[(size_t)e * 4 + 3];
    t += a0 * Wc[j] + a1 * Wc[f + j] + a2 * Wc[2 * f + j] + a3 * Wc[3 * f + j];
    size_t o = (size_t)d * f + j;
    atomicAdd(&S[o], t);
    atomicAdd(&SS[o], t * t);
    atomicMin(&MN[o], encf(t));
    atomicMax(&MX[o], encf(t));
    if (j == 0) atomicAdd(&DEG[d], 1u);
}

// ---------- finalize PNA stats; write bf16 Agg [mean|min|max|std] and the two PNA row scales ----------
__global__ void finalize_kernel(const float* __restrict__ P, const float* __restrict__ S,
                                const float* __restrict__ SS, const unsigned* __restrict__ MN,
                                const unsigned* __restrict__ MX, const unsigned* __restrict__ DEG,
                                const float* __restrict__ avgld,
                                __bf16* __restrict__ AGG, float* __restrict__ S1,
                                float* __restrict__ S2, int Nn, int f) {
    size_t i = (size_t)blockIdx.x * blockDim.x + threadIdx.x;
    if (i >= (size_t)Nn * f) return;
    int n = (int)(i / f);
    int j = (int)(i - (size_t)n * f);
    unsigned dg = DEG[n];
    float dd = (dg > 0) ? (float)dg : 1.f;
    float c = P[i], st = S[i], sst = SS[i];
    float mean = ((float)dg * c + st) / dd;
    float s2v  = (float)dg * c * c + 2.f * c * st + sst;
    float var  = s2v / dd - mean * mean;
    float stdv = sqrtf(fmaxf(var, 0.f) + EPSV);
    float mn = (dg > 0) ? (c + decf(MN[i])) : 0.f;
    float mx = (dg > 0) ? (c + decf(MX[i])) : 0.f;
    size_t b = (size_t)n * 4 * f + j;
    AGG[b]         = (__bf16)mean;
    AGG[b + f]     = (__bf16)mn;
    AGG[b + 2 * f] = (__bf16)mx;
    AGG[b + 3 * f] = (__bf16)stdv;
    if (j == 0) {
        float logd = logf(fmaxf((float)dg, 1.f) + 1.f);
        float av = avgld[0];
        S1[n] = logd / av;
        S2[n] = av / logd;
    }
}

// ---------- batch norm (one block per column) ----------
__global__ void bnstats_kernel(const float* __restrict__ L, float* __restrict__ mu,
                               float* __restrict__ var, int Nn) {
    __shared__ float ss[256], sq[256];
    int col = blockIdx.x, t = threadIdx.x;
    float a = 0.f, b = 0.f;
    for (int r = t; r < Nn; r += 256) {
        float v = L[(size_t)r * FEAT + col];
        a += v; b += v * v;
    }
    ss[t] = a; sq[t] = b;
    __syncthreads();
    for (int s = 128; s > 0; s >>= 1) {
        if (t < s) { ss[t] += ss[t + s]; sq[t] += sq[t + s]; }
        __syncthreads();
    }
    if (t == 0) {
        float m = ss[0] / (float)Nn;
        mu[col] = m;
        var[col] = sq[0] / (float)Nn - m * m;
    }
}
__global__ void bnapply_kernel(const float* __restrict__ L, const float* __restrict__ mu,
                               const float* __restrict__ var, const float* __restrict__ gamma,
                               const float* __restrict__ beta, float* __restrict__ H, int Nn) {
    size_t i = (size_t)blockIdx.x * blockDim.x + threadIdx.x;
    if (i >= (size_t)Nn * FEAT) return;
    int c = (int)(i % FEAT);
    float v = (L[i] - mu[c]) * rsqrtf(var[c] + EPSV) * gamma[c] + beta[c];
    H[i] = fmaxf(v, 0.f);
}

// ---------- global graph pooling ----------
__global__ void pool_kernel(const float* __restrict__ H, const int* __restrict__ batch,
                            unsigned* __restrict__ gmx, float* __restrict__ gsum,
                            float* __restrict__ gcnt, int Nn) {
    size_t i = (size_t)blockIdx.x * blockDim.x + threadIdx.x;
    if (i >= (size_t)Nn * FEAT) return;
    int n = (int)(i / FEAT), c = (int)(i % FEAT);
    int b = batch[n];
    float v = H[i];
    atomicMax(&gmx[(size_t)b * FEAT + c], encf(v));
    atomicAdd(&gsum[(size_t)b * FEAT + c], v);
    if (c == 0) atomicAdd(&gcnt[b], 1.f);
}
__global__ void poolbuild_kernel(const unsigned* __restrict__ gmx, const float* __restrict__ gsum,
                                 const float* __restrict__ gcnt, float* __restrict__ G) {
    int i = blockIdx.x * blockDim.x + threadIdx.x;
    if (i >= NGRAPH * 256) return;
    int b = i / 256, c = i % 256;
    float cnt = gcnt[b];
    if (c < FEAT)
        G[i] = (cnt > 0.f) ? decf(gmx[(size_t)b * FEAT + c]) : 0.f;
    else
        G[i] = gsum[(size_t)b * FEAT + (c - FEAT)] / fmaxf(cnt, 1.f);
}

// ---------- final tiny layer 64 -> 2 ----------
__global__ void lin3_kernel(const float* __restrict__ M2, const float* __restrict__ W3,
                            const float* __restrict__ b3, float* __restrict__ out) {
    int t = threadIdx.x;
    if (t >= NGRAPH * 2) return;
    int r = t >> 1, c = t & 1;
    float acc = b3[c];
    for (int k = 0; k < 64; ++k) acc += M2[(size_t)r * 64 + k] * W3[(size_t)k * 2 + c];
    out[(size_t)r * 2 + c] = acc;
}

// ---------- host-side GEMM dispatch (pre-swizzle W, then WMMA GEMM) ----------
static inline void run_gemm_f32(const float* A, const float* W, __bf16* Wsw,
                                const float* bias, const float* rs, float* C,
                                int M, int K, int Ncols, int flags, hipStream_t st) {
    int NT = Ncols / 16;
    int tot = K * Ncols;
    swizzlew_kernel<<<(tot + 255) / 256, 256, 0, st>>>(W, Wsw, K, NT);
    dim3 g((M + 127) / 128);
    if (Ncols == 32)      gemm_wmma<float, 2><<<g, 256, 0, st>>>(A, Wsw, bias, rs, C, M, K, flags);
    else if (Ncols == 64) gemm_wmma<float, 4><<<g, 256, 0, st>>>(A, Wsw, bias, rs, C, M, K, flags);
    else                  gemm_wmma<float, 8><<<g, 256, 0, st>>>(A, Wsw, bias, rs, C, M, K, flags);
}
static inline void run_gemm_bf16(const __bf16* A, const float* W, __bf16* Wsw,
                                 const float* rs, float* C, int M, int K, hipStream_t st) {
    int tot = K * FEAT;
    swizzlew_kernel<<<(tot + 255) / 256, 256, 0, st>>>(W, Wsw, K, 8);
    dim3 g((M + 127) / 128);
    gemm_wmma<__bf16, 8><<<g, 256, 0, st>>>(A, Wsw, nullptr, rs, C, M, K, 1 /*accumulate*/);
}

extern "C" void kernel_launch(void* const* d_in, const int* in_sizes, int n_in,
                              void* d_out, int out_size, void* d_ws, size_t ws_size,
                              hipStream_t stream) {
    const int Nn = NNODES, E = NEDGES;
    // ---- inputs (setup_inputs() dict order, nested dicts in insertion order) ----
    const float* x     = (const float*)d_in[0];
    const int*   ei    = (const int*)d_in[1];            // [2,E]: src row then dst row
    const float* ea    = (const float*)d_in[2];
    const int*   batch = (const int*)d_in[4];
    const float* avgld = (const float*)d_in[5];
    const float* W1 = (const float*)d_in[46]; const float* b1 = (const float*)d_in[47];
    const float* W2 = (const float*)d_in[48]; const float* b2 = (const float*)d_in[49];
    const float* W3 = (const float*)d_in[50]; const float* b3 = (const float*)d_in[51];

    // ---- carve workspace ----
    size_t off = 0;
    auto carve = [&](size_t bytes) {
        void* p = (char*)d_ws + off;
        off += (bytes + 255) & ~(size_t)255;
        return p;
    };
    float*    HA   = (float*)carve((size_t)Nn * FEAT * 4);
    float*    HB   = (float*)carve((size_t)Nn * FEAT * 4);
    float*    P    = (float*)carve((size_t)Nn * FEAT * 4);   // reused as post output O
    float*    Q    = (float*)carve((size_t)Nn * FEAT * 4);   // reused as lin output L
    float*    S    = (float*)carve((size_t)Nn * FEAT * 4);
    float*    SS   = (float*)carve((size_t)Nn * FEAT * 4);
    unsigned* MN   = (unsigned*)carve((size_t)Nn * FEAT * 4);
    unsigned* MX   = (unsigned*)carve((size_t)Nn * FEAT * 4);
    __bf16*   AGG  = (__bf16*)carve((size_t)Nn * 4 * FEAT * 2);
    unsigned* DEG  = (unsigned*)carve((size_t)Nn * 4);
    float*    S1   = (float*)carve((size_t)Nn * 4);
    float*    S2   = (float*)carve((size_t)Nn * 4);
    float*    Wc   = (float*)carve(4 * FEAT * 4);
    float*    bcv  = (float*)carve(FEAT * 4);
    float*    mu   = (float*)carve(FEAT * 4);
    float*    var  = (float*)carve(FEAT * 4);
    __bf16*   Wsw  = (__bf16*)carve((size_t)512 * FEAT * 2); // swizzled-weight staging (max K=512)
    unsigned* gmx  = (unsigned*)carve((size_t)NGRAPH * FEAT * 4);
    float*    gsum = (float*)carve((size_t)NGRAPH * FEAT * 4);
    float*    gcnt = (float*)carve(NGRAPH * 4);
    float*    G    = (float*)carve((size_t)NGRAPH * 256 * 4);
    float*    M1b  = (float*)carve((size_t)NGRAPH * FEAT * 4);
    float*    M2b  = (float*)carve((size_t)NGRAPH * 64 * 4);
    (void)ws_size; (void)in_sizes; (void)n_in; (void)out_size;

    const int* src = ei;
    const int* dst = ei + E;

    for (int l = 0; l < 4; ++l) {
        const int base = 6 + l * 10;
        const float* We    = (const float*)d_in[base + 0];
        const float* be    = (const float*)d_in[base + 1];
        const float* Wpre  = (const float*)d_in[base + 2];
        const float* bpre  = (const float*)d_in[base + 3];
        const float* Wpost = (const float*)d_in[base + 4];
        const float* bpost = (const float*)d_in[base + 5];
        const float* Wlin  = (const float*)d_in[base + 6];
        const float* blin  = (const float*)d_in[base + 7];
        const float* gamma = (const float*)d_in[base + 8];
        const float* beta  = (const float*)d_in[base + 9];

        const int f = (l == 0) ? 32 : FEAT;                  // f_in for this layer
        const float* Hin = (l == 0) ? x : ((l & 1) ? HB : HA);
        float* Hout = (l & 1) ? HA : HB;

        // fold edge encoder into pre_nn
        wc_kernel<<<1, f, 0, stream>>>(We, be, Wpre, bpre, Wc, bcv, f);

        // P = Hin@Wi + (be@We2 + bpre) ; Q = Hin@Wj
        run_gemm_f32(Hin, Wpre, Wsw, bcv, nullptr, P, Nn, f, f, 0, stream);
        run_gemm_f32(Hin, Wpre + (size_t)f * f, Wsw, nullptr, nullptr, Q, Nn, f, f, 0, stream);

        // init accumulators (live in L2: 4 * N * f * 4B <= 102 MB < 192 MB)
        long long nf = (long long)Nn * f;
        int gb = (int)((nf + 255) / 256);
        initf_kernel<<<gb, 256, 0, stream>>>(S, 0.f, nf);
        initf_kernel<<<gb, 256, 0, stream>>>(SS, 0.f, nf);
        initu_kernel<<<gb, 256, 0, stream>>>(MN, 0xFFFFFFFFu, nf);
        initu_kernel<<<gb, 256, 0, stream>>>(MX, 0u, nf);
        initu_kernel<<<(Nn + 255) / 256, 256, 0, stream>>>(DEG, 0u, Nn);

        // per-edge atomic aggregation
        long long ef = (long long)E * f;
        edge_kernel<<<(int)((ef + 255) / 256), 256, 0, stream>>>(
            src, dst, ea, Q, Wc, S, SS, MN, MX, DEG, E, f);

        // finalize PNA stats -> bf16 Agg + degree scalers
        finalize_kernel<<<gb, 256, 0, stream>>>(P, S, SS, MN, MX, DEG, avgld, AGG, S1, S2, Nn, f);

        // post_nn as 4 accumulating WMMA GEMMs into O(=P):
        float* O = P;
        run_gemm_f32(Hin, Wpost, Wsw, bpost, nullptr, O, Nn, f, FEAT, 0, stream);   // x part
        run_gemm_bf16(AGG, Wpost + (size_t)f * FEAT,     Wsw, nullptr, O, Nn, 4 * f, stream);
        run_gemm_bf16(AGG, Wpost + (size_t)5 * f * FEAT, Wsw, S1,      O, Nn, 4 * f, stream);
        run_gemm_bf16(AGG, Wpost + (size_t)9 * f * FEAT, Wsw, S2,      O, Nn, 4 * f, stream);

        // final conv linear, then BN + ReLU
        float* L = Q;
        run_gemm_f32(O, Wlin, Wsw, blin, nullptr, L, Nn, FEAT, FEAT, 0, stream);
        bnstats_kernel<<<FEAT, 256, 0, stream>>>(L, mu, var, Nn);
        bnapply_kernel<<<(int)(((long long)Nn * FEAT + 255) / 256), 256, 0, stream>>>(
            L, mu, var, gamma, beta, Hout, Nn);
    }

    const float* H = HA;  // layer 3 (l&1 == 1) wrote HA

    // ---- graph pooling ----
    initu_kernel<<<(NGRAPH * FEAT + 255) / 256, 256, 0, stream>>>(gmx, 0u, NGRAPH * FEAT);
    initf_kernel<<<(NGRAPH * FEAT + 255) / 256, 256, 0, stream>>>(gsum, 0.f, NGRAPH * FEAT);
    initf_kernel<<<1, 256, 0, stream>>>(gcnt, 0.f, NGRAPH);
    pool_kernel<<<(int)(((long long)Nn * FEAT + 255) / 256), 256, 0, stream>>>(
        H, batch, gmx, gsum, gcnt, Nn);
    poolbuild_kernel<<<(NGRAPH * 256 + 255) / 256, 256, 0, stream>>>(gmx, gsum, gcnt, G);

    // ---- MLP head (WMMA) ----
    run_gemm_f32(G,   W1, Wsw, b1, nullptr, M1b, NGRAPH, 256, FEAT, 2 /*relu*/, stream);
    run_gemm_f32(M1b, W2, Wsw, b2, nullptr, M2b, NGRAPH, FEAT, 64, 2 /*relu*/, stream);
    lin3_kernel<<<1, 256, 0, stream>>>(M2b, W3, b3, (float*)d_out);
}
